// CSLRTransformerBaseline_87780541596421
// MI455X (gfx1250) — matrix operand order repaired and checked by reading, not compiled, not run on hardware
//
#include <hip/hip_runtime.h>
#include <hip/hip_bf16.h>

// ---------------------------------------------------------------------------
// CSLR transformer forward for MI455X (gfx1250), wave32 + WMMA.
// All GEMMs go through v_wmma_f32_16x16x32_f16 (f16 in, f32 accumulate).
// Staging is branch-free: all operand tensors are padded (in workspace) so
// K % 32 == 0, lda/ldb % 4 == 0, and A/B tile reads never leave their
// allocations; only the epilogue store is masked. The expert out-projection
// GEMM additionally stages its A tiles with the Tensor Data Mover (TDM).
// ---------------------------------------------------------------------------

#define B_  32
#define T_  512
#define JD_ 86          // J*D2 = 43*2
#define JDP 96          // padded K for input projection
#define DM_ 512
#define H_  8
#define DH_ 64
#define FF_ 2048
#define E_  101
#define NC_ 101
#define NCP 128         // padded head rows
#define TP_ 128         // T/4 after two pools

typedef __attribute__((ext_vector_type(16))) _Float16 v16h;
typedef __attribute__((ext_vector_type(4)))  _Float16 v4h;
typedef __attribute__((ext_vector_type(8)))  float    v8f;
typedef __attribute__((ext_vector_type(4)))  unsigned int u32x4;
typedef __attribute__((ext_vector_type(8)))  int      i32x8;
typedef __attribute__((ext_vector_type(4)))  int      i32x4;

__device__ __forceinline__ float gelu_exact(float x) {
    return 0.5f * x * (1.0f + erff(x * 0.70710678118654752f));
}

// ---------------------------------------------------------------------------
// Generic batched GEMM:  C = act(alpha * A @ op(B) + bias)
//   TRANSB=true : Bm is [N,K] row-major (weights / K-matrix)  -> C = A @ Bm^T
//   TRANSB=false: Bm is [K,N] row-major                        -> C = A @ Bm
// Preconditions (guaranteed by the host-side padding):
//   K % 32 == 0; lda % 4 == 0; (TRANSB: ldb % 4 == 0)
//   A readable for ceil64(M) rows per batch slice; Bm readable for
//   ceil64-tile columns/rows touched by the grid.
// Block = 128 threads = 4 waves (2x2), block tile 64x64, each wave 32x32 via
// 2x2 v_wmma_f32_16x16x32_f16 fragments. Fragment layouts per ISA 7.12.2.
// ---------------------------------------------------------------------------
template<bool TRANSB, int ACT>
__global__ __launch_bounds__(128)
void gemm_wmma(const float* __restrict__ A, const float* __restrict__ Bm,
               const float* __restrict__ bias, float* __restrict__ C,
               int M, int N, int K, int lda, int ldb, int ldc,
               int innerB,
               long long sA1, long long sB1, long long sC1,
               long long sA0, long long sB0, long long sBias0, long long sC0,
               float alpha)
{
    // stride 40 halves = 80B: rows land on distinct banks for the 16-row
    // fragment reads; 80r + 2*kk (kk in {0,16}) is 8B aligned for v4h stores.
    __shared__ _Float16 As[64][40];
    __shared__ _Float16 Bs[64][40];

    const int z  = blockIdx.z;
    const int zo = z / innerB, zi = z % innerB;
    A  += zo * sA0 + zi * sA1;
    Bm += zo * sB0 + zi * sB1;
    C  += zo * sC0 + zi * sC1;
    if (bias) bias += zo * sBias0;

    const int tid  = threadIdx.x;
    const int lane = tid & 31;
    const int wave = tid >> 5;
    const int wm   = (wave >> 1) * 32;   // wave tile origin in block tile
    const int wn   = (wave & 1) * 32;
    const int m0   = blockIdx.y * 64;
    const int n0   = blockIdx.x * 64;

    v8f acc[2][2] = {};

    const int fm  = lane & 15;          // M (A) or N (B) index inside 16
    const int k8  = (lane >> 4) * 8;    // A: K base of lane half
    const int k16 = (lane >> 4) * 16;   // unused (B fragments also use k8)

    const int r  = tid >> 1;            // staging row (0..63)
    const int kk = (tid & 1) * 16;      // staging K sub-offset

    const float* aRow = A + (long long)(m0 + r) * lda + kk;
    const float* bRow = TRANSB ? (Bm + (long long)(n0 + r) * ldb + kk)
                               : (Bm + (long long)kk * ldb + (n0 + r));

    for (int k0 = 0; k0 < K; k0 += 32) {
        // ---- stage A tile 64x32 (f32 -> f16), branch-free ----
        {
            const float4* src = (const float4*)(aRow + k0);
            _Float16 h[16];
#pragma unroll
            for (int j = 0; j < 4; ++j) {
                const float4 f = src[j];
                h[4*j+0] = (_Float16)f.x; h[4*j+1] = (_Float16)f.y;
                h[4*j+2] = (_Float16)f.z; h[4*j+3] = (_Float16)f.w;
            }
            if (k0 + 32 < K)   // prefetch next K tile
                __builtin_prefetch(aRow + k0 + 32, 0, 1);
            v4h* dst = (v4h*)&As[r][kk];
#pragma unroll
            for (int j = 0; j < 4; ++j) dst[j] = *(const v4h*)&h[4*j];
        }
        // ---- stage B tile as [n][k] 64x32, branch-free ----
        {
            _Float16 h[16];
            if (TRANSB) {
                const float4* src = (const float4*)(bRow + k0);
#pragma unroll
                for (int j = 0; j < 4; ++j) {
                    const float4 f = src[j];
                    h[4*j+0] = (_Float16)f.x; h[4*j+1] = (_Float16)f.y;
                    h[4*j+2] = (_Float16)f.z; h[4*j+3] = (_Float16)f.w;
                }
            } else {
                const float* src = bRow + (long long)k0 * ldb;
#pragma unroll
                for (int i = 0; i < 16; ++i)
                    h[i] = (_Float16)src[(long long)i * ldb];
            }
            v4h* dst = (v4h*)&Bs[r][kk];
#pragma unroll
            for (int j = 0; j < 4; ++j) dst[j] = *(const v4h*)&h[4*j];
        }
        __syncthreads();

        // ---- build fragments and issue WMMAs ----
        v16h afrag[2], bfrag[2];
#pragma unroll
        for (int rr = 0; rr < 2; ++rr) {
            const int row = wm + rr * 16 + fm;
#pragma unroll
            for (int i = 0; i < 8; ++i) {
                afrag[rr][i]     = As[row][k8 + i];        // VGPR0..3: K = k8..k8+7
                afrag[rr][8 + i] = As[row][16 + k8 + i];   // VGPR4..7: K = 16+k8..
            }
        }
#pragma unroll
        for (int c = 0; c < 2; ++c) {
            const int col = wn + c * 16 + fm;
#pragma unroll
            for (int i = 0; i < 16; ++i)
                bfrag[c][i] = Bs[col][i < 8 ? k8 + i : 8 + k8 + i]; // K = k8..k8+7, then 16+k8..16+k8+7
        }
#pragma unroll
        for (int rr = 0; rr < 2; ++rr)
#pragma unroll
            for (int c = 0; c < 2; ++c)
                acc[rr][c] = __builtin_amdgcn_wmma_f32_16x16x32_f16(
                    false, afrag[rr], false, bfrag[c], (short)0, acc[rr][c], false, false);
        __syncthreads();
    }

    // ---- epilogue: alpha, bias, activation, masked f32 store ----
#pragma unroll
    for (int rr = 0; rr < 2; ++rr) {
#pragma unroll
        for (int c = 0; c < 2; ++c) {
#pragma unroll
            for (int j = 0; j < 8; ++j) {
                const int gm = m0 + wm + rr * 16 + (lane >> 4) * 8 + j;
                const int gn = n0 + wn + c * 16 + (lane & 15);
                if (gm < M && gn < N) {
                    float v = acc[rr][c][j] * alpha;
                    if (bias) v += bias[gn];
                    if (ACT == 1)      v = fmaxf(v, 0.0f);
                    else if (ACT == 2) v = gelu_exact(v);
                    C[(long long)gm * ldc + gn] = v;
                }
            }
        }
    }
}

// ---------------------------------------------------------------------------
// TDM-staged GEMM variant (TRANSB, no activation): the 64x32 f32 A tile is
// moved global->LDS by the Tensor Data Mover; fragments convert f32->f16 on
// load. D# per CDNA5 ISA §8.3/8.4:
//   group0: count=1 | lds_addr | global_addr[56:0] | type=2
//   group1: data_size=4B, pad_enable, pad_interval=32 dwords, pad_amount=4
//           dwords (-> LDS row stride 36 dwords, conflict-free fragment
//           reads), tensor_dim0=K, tensor_dim1=M (TDM zero-fills OOB rows),
//           tile_dim0=32, tile_dim1=64, tensor_dim0_stride=lda
//   groups 2/3 (and the trailing descriptor slot): zero (2D tensor)
// ---------------------------------------------------------------------------
__global__ __launch_bounds__(128)
void gemm_wmma_tdm(const float* __restrict__ A, const float* __restrict__ Bm,
                   const float* __restrict__ bias, float* __restrict__ C,
                   int M, int N, int K, int lda, int ldb, int ldc,
                   long long sA0, long long sB0, long long sBias0, long long sC0,
                   float alpha)
{
    __shared__ float    Af[64][36];     // f32 A tile staged by TDM (36-dword rows)
    __shared__ _Float16 Bs[64][40];

    const int z = blockIdx.z;
    A  += z * sA0;
    Bm += z * sB0;
    C  += z * sC0;
    if (bias) bias += z * sBias0;

    const int tid  = threadIdx.x;
    const int lane = tid & 31;
    const int wave = tid >> 5;
    const int wm   = (wave >> 1) * 32;
    const int wn   = (wave & 1) * 32;
    const int m0   = blockIdx.y * 64;
    const int n0   = blockIdx.x * 64;

    v8f acc[2][2] = {};
    const int fm  = lane & 15;
    const int k8  = (lane >> 4) * 8;
    const int k16 = (lane >> 4) * 16;
    const int r   = tid >> 1;
    const int kk  = (tid & 1) * 16;

    const float* bRow = Bm + (long long)(n0 + r) * ldb + kk;
    const unsigned int ldsA = (unsigned int)(unsigned long long)&Af[0][0];

    for (int k0 = 0; k0 < K; k0 += 32) {
        // ---- stage B tile as [n][k] (f32 -> f16), branch-free ----
        {
            const float4* src = (const float4*)(bRow + k0);
            _Float16 h[16];
#pragma unroll
            for (int j = 0; j < 4; ++j) {
                const float4 f = src[j];
                h[4*j+0] = (_Float16)f.x; h[4*j+1] = (_Float16)f.y;
                h[4*j+2] = (_Float16)f.z; h[4*j+3] = (_Float16)f.w;
            }
            v4h* dst = (v4h*)&Bs[r][kk];
#pragma unroll
            for (int j = 0; j < 4; ++j) dst[j] = *(const v4h*)&h[4*j];
        }
        // ---- TDM: wave 0 issues the async 64x32 f32 tile load ----
        if (wave == 0) {
            const unsigned long long ga =
                (unsigned long long)(const void*)(A + (long long)m0 * lda + k0);
            u32x4 g0;
            g0[0] = 1u;                                   // count=1 (valid user D#)
            g0[1] = ldsA;                                 // lds_addr (bytes)
            g0[2] = (unsigned int)ga;                     // global_addr[31:0]
            g0[3] = (unsigned int)((ga >> 32) & 0x01FFFFFFu) | 0x80000000u; // addr[56:32] | type=2
            i32x8 g1;
            g1[0] = (int)((2u << 16) | (1u << 20) | (4u << 22) | (3u << 25));
                     // data_size=4B | pad_enable | pad_interval=32dw | pad_amount=4dw
            g1[1] = (int)(((unsigned)K & 0xffffu) << 16);                    // tensor_dim0 lo
            g1[2] = (int)((((unsigned)K >> 16) & 0xffffu) |
                          (((unsigned)M & 0xffffu) << 16));                  // dim0 hi | dim1 lo
            g1[3] = (int)((((unsigned)M >> 16) & 0xffffu) | (32u << 16));    // dim1 hi | tile_dim0=32
            g1[4] = 64;                                                      // tile_dim1=64
            g1[5] = lda;                                                     // dim0 stride lo
            g1[6] = 0;
            g1[7] = 0;
            i32x4 g2 = {};
            i32x4 g3 = {};
            i32x8 g4 = {};
            __builtin_amdgcn_tensor_load_to_lds(g0, g1, g2, g3, g4, 0);
            __builtin_amdgcn_s_wait_tensorcnt(0);
        }
        __syncthreads();

        // ---- fragments: A from f32 LDS (convert), B from f16 LDS ----
        v16h afrag[2], bfrag[2];
#pragma unroll
        for (int rr = 0; rr < 2; ++rr) {
            const int row = wm + rr * 16 + fm;
            const float4 f0 = *(const float4*)&Af[row][k8];
            const float4 f1 = *(const float4*)&Af[row][k8 + 4];
            const float4 f2 = *(const float4*)&Af[row][16 + k8];
            const float4 f3 = *(const float4*)&Af[row][16 + k8 + 4];
            afrag[rr][0]  = (_Float16)f0.x; afrag[rr][1]  = (_Float16)f0.y;
            afrag[rr][2]  = (_Float16)f0.z; afrag[rr][3]  = (_Float16)f0.w;
            afrag[rr][4]  = (_Float16)f1.x; afrag[rr][5]  = (_Float16)f1.y;
            afrag[rr][6]  = (_Float16)f1.z; afrag[rr][7]  = (_Float16)f1.w;
            afrag[rr][8]  = (_Float16)f2.x; afrag[rr][9]  = (_Float16)f2.y;
            afrag[rr][10] = (_Float16)f2.z; afrag[rr][11] = (_Float16)f2.w;
            afrag[rr][12] = (_Float16)f3.x; afrag[rr][13] = (_Float16)f3.y;
            afrag[rr][14] = (_Float16)f3.z; afrag[rr][15] = (_Float16)f3.w;
        }
#pragma unroll
        for (int c = 0; c < 2; ++c) {
            const int col = wn + c * 16 + fm;
#pragma unroll
            for (int i = 0; i < 16; ++i)
                bfrag[c][i] = Bs[col][i < 8 ? k8 + i : 8 + k8 + i];
        }
#pragma unroll
        for (int rr = 0; rr < 2; ++rr)
#pragma unroll
            for (int c = 0; c < 2; ++c)
                acc[rr][c] = __builtin_amdgcn_wmma_f32_16x16x32_f16(
                    false, afrag[rr], false, bfrag[c], (short)0, acc[rr][c], false, false);
        __syncthreads();
    }

#pragma unroll
    for (int rr = 0; rr < 2; ++rr) {
#pragma unroll
        for (int c = 0; c < 2; ++c) {
#pragma unroll
            for (int j = 0; j < 8; ++j) {
                const int gm = m0 + wm + rr * 16 + (lane >> 4) * 8 + j;
                const int gn = n0 + wn + c * 16 + (lane & 15);
                if (gm < M && gn < N) {
                    float v = acc[rr][c][j] * alpha;
                    if (bias) v += bias[gn];
                    C[(long long)gm * ldc + gn] = v;
                }
            }
        }
    }
}

// ---------------------------------------------------------------------------
// Elementwise / reduction kernels
// ---------------------------------------------------------------------------

// flat zero-padded copy: dst[r*kout + k] = (k < kin) ? src[r*kin + k] : 0
__global__ void pad_copy(const float* __restrict__ src, float* __restrict__ dst,
                         int kin, int kout, long long n)
{
    long long i = (long long)blockIdx.x * 256 + threadIdx.x;
    if (i >= n) return;
    const int k = (int)(i % kout);
    const long long r = i / kout;
    dst[i] = (k < kin) ? src[r * (long long)kin + k] : 0.0f;
}

// beff[e,n] = qkv_b[e,n] + sum_d pos[e,d] * qkv_w[e,n,d]   (n in [0,1536))
__global__ void expert_beff(const float* __restrict__ pos, const float* __restrict__ qkvw,
                            const float* __restrict__ qkvb, float* __restrict__ beff,
                            long long n)
{
    long long i = (long long)blockIdx.x * 256 + threadIdx.x;
    if (i >= n) return;
    const long long e = i / (3 * DM_);
    const float* p = pos + e * DM_;
    const float* w = qkvw + i * DM_;     // row (e,n) of [E*1536, 512]
    float s = qkvb[i];
    for (int d = 0; d < DM_; ++d) s += p[d] * w[d];
    beff[i] = s;
}

__global__ void add_posenc(float* __restrict__ x, long long n)
{
    long long i = (long long)blockIdx.x * 256 + threadIdx.x;
    if (i >= n) return;
    const int d = (int)(i % DM_);
    const int t = (int)((i / DM_) % T_);
    const int h2 = d & ~1;
    const float freq = __expf(-logf(10000.0f) * (float)h2 / (float)DM_);
    const float a = (float)t * freq;
    x[i] += (d & 1) ? cosf(a) : sinf(a);
}

// y = LN(x [+ res]) * g[row/group] + b[row/group] ; one 128-thread block per row
__global__ __launch_bounds__(128)
void ln_res(const float* __restrict__ x, const float* __restrict__ res,
            const float* __restrict__ g, const float* __restrict__ b,
            float* __restrict__ y, int D, int group)
{
    __shared__ float s1s[4], s2s[4];
    const int row = blockIdx.x, tid = threadIdx.x;
    const float* xr = x + (long long)row * D;
    const float* rr = res ? res + (long long)row * D : nullptr;
    float s1 = 0.0f, s2 = 0.0f;
    for (int i = tid; i < D; i += 128) {
        float v = xr[i] + (rr ? rr[i] : 0.0f);
        s1 += v; s2 += v * v;
    }
    for (int o = 16; o > 0; o >>= 1) { s1 += __shfl_xor(s1, o, 32); s2 += __shfl_xor(s2, o, 32); }
    if ((tid & 31) == 0) { s1s[tid >> 5] = s1; s2s[tid >> 5] = s2; }
    __syncthreads();
    s1 = s1s[0] + s1s[1] + s1s[2] + s1s[3];
    s2 = s2s[0] + s2s[1] + s2s[2] + s2s[3];
    const float mu   = s1 / (float)D;
    const float var  = s2 / (float)D - mu * mu;
    const float rstd = rsqrtf(var + 1e-5f);
    const float* gg = g + (long long)(row / group) * D;
    const float* bb = b + (long long)(row / group) * D;
    float* yr = y + (long long)row * D;
    for (int i = tid; i < D; i += 128) {
        float v = xr[i] + (rr ? rr[i] : 0.0f);
        yr[i] = (v - mu) * rstd * gg[i] + bb[i];
    }
}

__global__ __launch_bounds__(128)
void softmax_rows(float* __restrict__ x, int L)
{
    __shared__ float red[4];
    const int row = blockIdx.x, tid = threadIdx.x;
    float* xr = x + (long long)row * L;
    float m = -3.4e38f;
    for (int i = tid; i < L; i += 128) m = fmaxf(m, xr[i]);
    for (int o = 16; o > 0; o >>= 1) m = fmaxf(m, __shfl_xor(m, o, 32));
    if ((tid & 31) == 0) red[tid >> 5] = m;
    __syncthreads();
    m = fmaxf(fmaxf(red[0], red[1]), fmaxf(red[2], red[3]));
    __syncthreads();
    float s = 0.0f;
    for (int i = tid; i < L; i += 128) { float e = __expf(xr[i] - m); xr[i] = e; s += e; }
    for (int o = 16; o > 0; o >>= 1) s += __shfl_xor(s, o, 32);
    if ((tid & 31) == 0) red[tid >> 5] = s;
    __syncthreads();
    s = red[0] + red[1] + red[2] + red[3];
    const float inv = 1.0f / s;
    for (int i = tid; i < L; i += 128) xr[i] *= inv;
}

// x: [B, T, C] channels-last, w: [O, C, 3], y[b,t,o] = gelu(conv); pad=1
__global__ __launch_bounds__(128)
void conv3_gelu(const float* __restrict__ x, const float* __restrict__ w,
                const float* __restrict__ bias, float* __restrict__ y, int T)
{
    __shared__ float xs[3][DM_];
    const int t = blockIdx.x, b = blockIdx.y, tid = threadIdx.x;
    const float* xb = x + (long long)b * T * DM_;
    for (int k = 0; k < 3; ++k) {
        const int tt = t + k - 1;
        for (int i = tid; i < DM_; i += 128)
            xs[k][i] = (tt >= 0 && tt < T) ? xb[(long long)tt * DM_ + i] : 0.0f;
    }
    __syncthreads();
    for (int o = tid; o < DM_; o += 128) {
        const float* wo = w + (long long)o * DM_ * 3;
        float acc = bias[o];
        for (int i = 0; i < DM_; ++i)
            acc += xs[0][i] * wo[i*3+0] + xs[1][i] * wo[i*3+1] + xs[2][i] * wo[i*3+2];
        y[((long long)b * T + t) * DM_ + o] = gelu_exact(acc);
    }
}

// [B, 2*Tout, C] -> [B, Tout, C] averaging adjacent time steps
__global__ void avgpool2(const float* __restrict__ x, float* __restrict__ y,
                         int Tout, long long n)
{
    long long i = (long long)blockIdx.x * 256 + threadIdx.x;
    if (i >= n) return;
    const int c = (int)(i % DM_);
    const long long tp = (i / DM_) % Tout;
    const long long b  = i / ((long long)DM_ * Tout);
    const float* src = x + ((b * (2 * Tout) + 2 * tp) * DM_) + c;
    y[i] = 0.5f * (src[0] + src[DM_]);
}

// x: [E*B, 128, 512] -> y[E*B, 512] (mean over 128 positions)
__global__ void mean_t(const float* __restrict__ x, float* __restrict__ y, long long n)
{
    long long i = (long long)blockIdx.x * 256 + threadIdx.x;
    if (i >= n) return;
    const long long row = i / DM_;
    const int d = (int)(i % DM_);
    const float* src = x + row * (long long)TP_ * DM_ + d;
    float s = 0.0f;
    for (int t = 0; t < TP_; ++t) s += src[(long long)t * DM_];
    y[i] = s * (1.0f / (float)TP_);
}

// lg: [(e*B+b), NC] -> out: [B, E, NC]
__global__ void final_transpose(const float* __restrict__ lg, float* __restrict__ out, long long n)
{
    long long i = (long long)blockIdx.x * 256 + threadIdx.x;
    if (i >= n) return;
    const int c = (int)(i % NC_);
    const int e = (int)((i / NC_) % E_);
    const int b = (int)(i / ((long long)NC_ * E_));
    out[i] = lg[((long long)e * B_ + b) * NC_ + c];
}

// ---------------------------------------------------------------------------
// Host side
// ---------------------------------------------------------------------------
static void gemm(hipStream_t st, bool transb, int act,
                 const float* A, const float* Bm, const float* bias, float* C,
                 int M, int N, int K, int lda, int ldb, int ldc,
                 int batch, int innerB,
                 long long sA0, long long sB0, long long sBias0, long long sC0,
                 long long sA1, long long sB1, long long sC1,
                 float alpha)
{
    dim3 grid((N + 63) / 64, (M + 63) / 64, batch), blk(128);
#define GARGS A, Bm, bias, C, M, N, K, lda, ldb, ldc, innerB, sA1, sB1, sC1, sA0, sB0, sBias0, sC0, alpha
    if (transb) {
        if      (act == 0) gemm_wmma<true, 0><<<grid, blk, 0, st>>>(GARGS);
        else if (act == 1) gemm_wmma<true, 1><<<grid, blk, 0, st>>>(GARGS);
        else               gemm_wmma<true, 2><<<grid, blk, 0, st>>>(GARGS);
    } else {
        gemm_wmma<false, 0><<<grid, blk, 0, st>>>(GARGS);   // NN is only used without act
    }
#undef GARGS
}

static inline dim3 g1(long long n) { return dim3((unsigned)((n + 255) / 256)); }

extern "C" void kernel_launch(void* const* d_in, const int* in_sizes, int n_in,
                              void* d_out, int out_size, void* d_ws, size_t ws_size,
                              hipStream_t stream)
{
    (void)in_sizes; (void)n_in; (void)out_size; (void)ws_size;
    int ii = 0;
    const float* poses   = (const float*)d_in[ii++];
    const float* W_in    = (const float*)d_in[ii++];
    const float* b_in    = (const float*)d_in[ii++];
    const float* eq_w    = (const float*)d_in[ii++];
    const float* eq_b    = (const float*)d_in[ii++];
    const float* eo_w    = (const float*)d_in[ii++];
    const float* eo_b    = (const float*)d_in[ii++];
    const float* ef1_w   = (const float*)d_in[ii++];
    const float* ef1_b   = (const float*)d_in[ii++];
    const float* ef2_w   = (const float*)d_in[ii++];
    const float* ef2_b   = (const float*)d_in[ii++];
    const float* en1_g   = (const float*)d_in[ii++];
    const float* en1_b   = (const float*)d_in[ii++];
    const float* en2_g   = (const float*)d_in[ii++];
    const float* en2_b   = (const float*)d_in[ii++];
    const float* c1_w    = (const float*)d_in[ii++];
    const float* c1_b    = (const float*)d_in[ii++];
    const float* c2_w    = (const float*)d_in[ii++];
    const float* c2_b    = (const float*)d_in[ii++];
    const float* mlp_w   = (const float*)d_in[ii++];
    const float* mlp_b   = (const float*)d_in[ii++];
    const float* xp_pos  = (const float*)d_in[ii++];
    const float* xp_qkvw = (const float*)d_in[ii++];
    const float* xp_qkvb = (const float*)d_in[ii++];
    const float* xp_ow   = (const float*)d_in[ii++];
    const float* xp_ob   = (const float*)d_in[ii++];
    const float* xp_ng   = (const float*)d_in[ii++];
    const float* xp_nb   = (const float*)d_in[ii++];
    const float* xp_lw   = (const float*)d_in[ii++];
    const float* xp_lb   = (const float*)d_in[ii++];
    const float* fn_g    = (const float*)d_in[ii++];
    const float* fn_b    = (const float*)d_in[ii++];
    const float* hd_w    = (const float*)d_in[ii++];
    const float* hd_b    = (const float*)d_in[ii++];

    // ---- workspace plan (floats). Persistent region + shared scratch region:
    // encoder scratch and expert scratch overlap (encoder dead before experts).
    float* ws = (float*)d_ws;
    size_t off = 0;
    auto alloc = [&](long long nElem) { float* p = ws + off; off += (size_t)((nElem + 63) & ~63LL); return p; };

    const long long BT = (long long)B_ * T_;            // 16384
    float* membuf  = alloc((long long)B_ * TP_ * DM_);  // persistent: memory tensor
    float* beff    = alloc((long long)E_ * 3 * DM_);    // persistent: expert qkv effective bias
    float* hdw_pad = alloc((long long)NCP * DM_);       // persistent: head weight padded to 128 rows
    const size_t scratch0 = off;

    // encoder-phase scratch
    float* pospad = alloc(BT * JDP);                    // poses padded K 86 -> 96
    float* winpad = alloc((long long)DM_ * JDP);        // W_in padded K 86 -> 96
    float* xbuf   = alloc(BT * DM_);
    float* qkvbuf = alloc(BT * 3 * DM_);
    float* scores = alloc((long long)B_ * H_ * T_ * T_);
    float* ctxbuf = alloc(BT * DM_);
    float* tmpbuf = alloc(BT * DM_);
    float* ffbuf  = alloc(BT * FF_);
    float* cb1    = alloc(BT * DM_);
    float* pb1    = alloc((long long)B_ * 256 * DM_);
    float* cb2    = alloc((long long)B_ * 256 * DM_);
    float* pb2    = alloc((long long)B_ * TP_ * DM_);

    // expert-phase scratch (reuses encoder region). cmean/clbuf over-allocated
    // by 64 rows so A-tile loads past logical M stay in-bounds (results masked).
    off = scratch0;
    float* xqkv  = alloc((long long)E_ * B_ * TP_ * 3 * DM_);
    float* xs    = alloc((long long)E_ * B_ * TP_ * TP_);
    float* xctx  = alloc((long long)E_ * B_ * TP_ * DM_);
    float* xoutb = alloc((long long)E_ * B_ * TP_ * DM_);
    float* cmean = alloc(((long long)E_ * B_ + 64) * DM_);
    float* clbuf = alloc(((long long)E_ * B_ + 64) * DM_);
    float* lgbuf = alloc((long long)E_ * B_ * NC_);

    const hipStream_t st = stream;

    // ---- pad inputs for branch-free GEMM staging ----
    pad_copy<<<g1(BT * JDP), 256, 0, st>>>(poses, pospad, JD_, JDP, BT * JDP);
    pad_copy<<<g1((long long)DM_ * JDP), 256, 0, st>>>(W_in, winpad, JD_, JDP, (long long)DM_ * JDP);
    pad_copy<<<g1((long long)NCP * DM_), 256, 0, st>>>(hd_w, hdw_pad,
                                                       NC_ * DM_, NCP * DM_, (long long)NCP * DM_);

    // ---- input projection + positional encoding ----
    gemm(st, true, 0, pospad, winpad, b_in, xbuf,
         (int)BT, DM_, JDP, JDP, JDP, DM_, 1, 1, 0,0,0,0, 0,0,0, 1.0f);
    add_posenc<<<g1(BT * DM_), 256, 0, st>>>(xbuf, BT * DM_);

    // ---- 2 encoder layers (post-norm, relu FFN) ----
    for (int l = 0; l < 2; ++l) {
        const float* qw  = eq_w  + (long long)l * 3 * DM_ * DM_;
        const float* qb  = eq_b  + (long long)l * 3 * DM_;
        const float* ow  = eo_w  + (long long)l * DM_ * DM_;
        const float* ob  = eo_b  + (long long)l * DM_;
        const float* f1w = ef1_w + (long long)l * FF_ * DM_;
        const float* f1b = ef1_b + (long long)l * FF_;
        const float* f2w = ef2_w + (long long)l * DM_ * FF_;
        const float* f2b = ef2_b + (long long)l * DM_;

        // qkv: [BT, 1536]
        gemm(st, true, 0, xbuf, qw, qb, qkvbuf,
             (int)BT, 3*DM_, DM_, DM_, DM_, 3*DM_, 1, 1, 0,0,0,0, 0,0,0, 1.0f);
        // scores[b,h] = Q @ K^T / 8 ; batch z = b*H + h
        gemm(st, true, 0, qkvbuf, qkvbuf + DM_, nullptr, scores,
             T_, T_, DH_, 3*DM_, 3*DM_, T_, B_*H_, H_,
             (long long)T_*3*DM_, (long long)T_*3*DM_, 0, (long long)H_*T_*T_,
             DH_, DH_, (long long)T_*T_, 0.125f);
        softmax_rows<<<dim3(B_*H_*T_), 128, 0, st>>>(scores, T_);
        // ctx[b,h] = P @ V  (NN), scatter heads into [B,T,DM]
        gemm(st, false, 0, scores, qkvbuf + 2*DM_, nullptr, ctxbuf,
             T_, DH_, T_, T_, 3*DM_, DM_, B_*H_, H_,
             (long long)H_*T_*T_, (long long)T_*3*DM_, 0, (long long)T_*DM_,
             (long long)T_*T_, DH_, DH_, 1.0f);
        // out proj + residual LN
        gemm(st, true, 0, ctxbuf, ow, ob, tmpbuf,
             (int)BT, DM_, DM_, DM_, DM_, DM_, 1, 1, 0,0,0,0, 0,0,0, 1.0f);
        ln_res<<<dim3((unsigned)BT), 128, 0, st>>>(xbuf, tmpbuf,
             en1_g + (long long)l*DM_, en1_b + (long long)l*DM_, xbuf, DM_, (int)BT);
        // FFN
        gemm(st, true, 1, xbuf, f1w, f1b, ffbuf,
             (int)BT, FF_, DM_, DM_, DM_, FF_, 1, 1, 0,0,0,0, 0,0,0, 1.0f);
        gemm(st, true, 0, ffbuf, f2w, f2b, tmpbuf,
             (int)BT, DM_, FF_, FF_, FF_, DM_, 1, 1, 0,0,0,0, 0,0,0, 1.0f);
        ln_res<<<dim3((unsigned)BT), 128, 0, st>>>(xbuf, tmpbuf,
             en2_g + (long long)l*DM_, en2_b + (long long)l*DM_, xbuf, DM_, (int)BT);
    }

    // ---- temporal pooling: conv-gelu-pool x2, then gelu MLP -> memory ----
    conv3_gelu<<<dim3(T_, B_), 128, 0, st>>>(xbuf, c1_w, c1_b, cb1, T_);
    avgpool2<<<g1((long long)B_*256*DM_), 256, 0, st>>>(cb1, pb1, 256, (long long)B_*256*DM_);
    conv3_gelu<<<dim3(256, B_), 128, 0, st>>>(pb1, c2_w, c2_b, cb2, 256);
    avgpool2<<<g1((long long)B_*TP_*DM_), 256, 0, st>>>(cb2, pb2, TP_, (long long)B_*TP_*DM_);
    gemm(st, true, 2, pb2, mlp_w, mlp_b, membuf,
         B_*TP_, DM_, DM_, DM_, DM_, DM_, 1, 1, 0,0,0,0, 0,0,0, 1.0f);

    // ---- experts: fold +pos into effective qkv bias (VALU dot kernel) ----
    expert_beff<<<g1((long long)E_*3*DM_), 256, 0, st>>>(xp_pos, xp_qkvw, xp_qkvb, beff,
                                                        (long long)E_*3*DM_);
    // qkv for all experts: A = shared memory tensor (outer stride 0)
    gemm(st, true, 0, membuf, xp_qkvw, beff, xqkv,
         B_*TP_, 3*DM_, DM_, DM_, DM_, 3*DM_, E_, 1,
         0, (long long)3*DM_*DM_, (long long)3*DM_, (long long)B_*TP_*3*DM_, 0,0,0, 1.0f);
    // scores per (e,b): 128x128, K=512, scale 1/sqrt(512)
    gemm(st, true, 0, xqkv, xqkv + DM_, nullptr, xs,
         TP_, TP_, DM_, 3*DM_, 3*DM_, TP_, E_*B_, 1,
         (long long)TP_*3*DM_, (long long)TP_*3*DM_, 0, (long long)TP_*TP_,
         0,0,0, 0.04419417382415922f);
    softmax_rows<<<dim3(E_*B_*TP_), 128, 0, st>>>(xs, TP_);
    // ctx = P @ V (NN)
    gemm(st, false, 0, xs, xqkv + 2*DM_, nullptr, xctx,
         TP_, DM_, TP_, TP_, 3*DM_, DM_, E_*B_, 1,
         (long long)TP_*TP_, (long long)TP_*3*DM_, 0, (long long)TP_*DM_,
         0,0,0, 1.0f);
    // out proj per expert -- TDM-staged GEMM (A tiles via tensor_load_to_lds)
    {
        dim3 grid((DM_ + 63) / 64, (B_ * TP_ + 63) / 64, E_);
        gemm_wmma_tdm<<<grid, 128, 0, st>>>(xctx, xp_ow, xp_ob, xoutb,
            B_*TP_, DM_, DM_, DM_, DM_, DM_,
            (long long)B_*TP_*DM_, (long long)DM_*DM_, DM_, (long long)B_*TP_*DM_, 1.0f);
    }
    // mean over positions -> [E*B, 512], expert LN (per-expert gamma/beta), linear
    mean_t<<<g1((long long)E_*B_*DM_), 256, 0, st>>>(xoutb, cmean, (long long)E_*B_*DM_);
    ln_res<<<dim3(E_*B_), 128, 0, st>>>(cmean, nullptr, xp_ng, xp_nb, cmean, DM_, B_);
    gemm(st, true, 0, cmean, xp_lw, xp_lb, clbuf,
         B_, DM_, DM_, DM_, DM_, DM_, E_, 1,
         (long long)B_*DM_, (long long)DM_*DM_, DM_, (long long)B_*DM_, 0,0,0, 1.0f);
    // final LN (shared) + head (padded weights) + transpose to [B, E, NC]
    ln_res<<<dim3(E_*B_), 128, 0, st>>>(clbuf, nullptr, fn_g, fn_b, clbuf, DM_, E_*B_);
    gemm(st, true, 0, clbuf, hdw_pad, hd_b, lgbuf,
         E_*B_, NC_, DM_, DM_, DM_, NC_, 1, 1, 0,0,0,0, 0,0,0, 1.0f);
    final_transpose<<<g1((long long)B_*E_*NC_), 256, 0, st>>>(lgbuf, (float*)d_out,
                                                             (long long)B_*E_*NC_);
}
